// HybridReservoirBlock_48636209660446
// MI455X (gfx1250) — compile-verified, matrix operand order checked by reading, not run on hardware
//
#include <hip/hip_runtime.h>

typedef __bf16 bf16;
typedef __bf16 v16bf __attribute__((ext_vector_type(16)));
typedef __bf16 v8bf  __attribute__((ext_vector_type(8)));
typedef float  v8f   __attribute__((ext_vector_type(8)));
typedef float  v4f   __attribute__((ext_vector_type(4)));
typedef unsigned v4u __attribute__((ext_vector_type(4)));
typedef int      v4i __attribute__((ext_vector_type(4)));
typedef int      v8i __attribute__((ext_vector_type(8)));

typedef __attribute__((address_space(3))) bf16 lbf16;
typedef __attribute__((address_space(3))) void lvoid;
typedef __attribute__((address_space(1))) void gvoid;

// GCC-style int4 vector matching the async-LDS builtin's pointee type exactly
typedef int v4i_g __attribute__((vector_size(16)));
typedef __attribute__((address_space(1))) v4i_g g_v4i;
typedef __attribute__((address_space(3))) v4i_g l_v4i;

#define BATCH 8
#define SEQ   2048
#define EDIM  1024
#define HDIM  1024
#define ROWS  (BATCH*SEQ)

#define GEMM_ASYNC (__has_builtin(__builtin_amdgcn_global_load_async_to_lds_b128) && \
                    __has_builtin(__builtin_amdgcn_s_wait_asynccnt))
#define HAVE_TDM   (__has_builtin(__builtin_amdgcn_tensor_load_to_lds) && \
                    __has_builtin(__builtin_amdgcn_s_wait_tensorcnt))

__device__ inline bf16 f2bf(float f) {
  union { float f; unsigned u; } cv; cv.f = f;
  unsigned r = cv.u + 0x7FFFu + ((cv.u >> 16) & 1u);   // round-to-nearest-even
  union { unsigned short s; bf16 b; } o; o.s = (unsigned short)(r >> 16);
  return o.b;
}

union FragBF { v16bf v; v8bf h[2]; };

__device__ inline unsigned lds_addr_of(const bf16* p) {
  return (unsigned)(unsigned long)(lbf16*)p;
}

#if GEMM_ASYNC
__device__ inline void async_ld_b128(const bf16* g, bf16* l) {
  __builtin_amdgcn_global_load_async_to_lds_b128(
      (g_v4i*)(gvoid*)g, (l_v4i*)(lvoid*)l, 0, 0);
}
#endif

#if HAVE_TDM
// 2D TDM tile load: tile_d1 rows of tile_d0 elements (bf16), row stride
// stride0 elements; optional LDS pad of +4 DWORDs every 16 DWORDs (LDT=40).
__device__ inline void tdm_load_2d(const bf16* gsrc, unsigned lds_byte,
                                   unsigned tile_d0, unsigned tile_d1,
                                   unsigned long stride0, unsigned tens_d0,
                                   unsigned tens_d1, bool pad)
{
  union { v4u v; unsigned a[4]; } g0 = {};
  union { v8i v; unsigned a[8]; } g1 = {};
  union { v4i v; int a[4]; } gz = {};
  unsigned long ga = (unsigned long)gsrc;
  g0.a[0] = 1u;                                             // count=1 (valid user D#)
  g0.a[1] = lds_byte;                                       // lds_addr
  g0.a[2] = (unsigned)ga;                                   // global_addr[31:0]
  g0.a[3] = (unsigned)((ga >> 32) & 0x01ffffffu) | (2u << 30); // addr[56:32] | type=2
  unsigned w0 = (1u << 16);                                 // data_size = 2 bytes
  if (pad) w0 |= (1u << 20) | (3u << 22) | (3u << 25);      // pad: every 16 DW add 4 DW
  g1.a[0] = w0;
  g1.a[1] = (tens_d0 & 0xffffu) << 16;                      // tensor_dim0[15:0]
  g1.a[2] = ((tens_d0 >> 16) & 0xffffu) | ((tens_d1 & 0xffffu) << 16);
  g1.a[3] = ((tens_d1 >> 16) & 0xffffu) | ((tile_d0 & 0xffffu) << 16);
  g1.a[4] = (tile_d1 & 0xffffu);                            // tile_dim1 (tile_dim2=0)
  g1.a[5] = (unsigned)(stride0 & 0xffffffffu);              // tensor_dim0_stride
  g1.a[6] = (unsigned)((stride0 >> 32) & 0xffffu);
  g1.a[7] = 0;
#if __clang_major__ >= 23
  union { v8i v; int a[8]; } gz8 = {};
  __builtin_amdgcn_tensor_load_to_lds(g0.v, g1.v, gz.v, gz.v, gz8.v, 0);
#else
  __builtin_amdgcn_tensor_load_to_lds(g0.v, g1.v, gz.v, gz.v, 0);
#endif
}
#endif

// ---------------------------------------------------------------- casts
__global__ __launch_bounds__(256) void k_f32_to_bf16(
    const float* __restrict__ src, bf16* __restrict__ dst, int n)
{
  int i = (blockIdx.x * 256 + threadIdx.x) * 4;
  if (i + 3 < n) {
    v4f v = *(const v4f*)(src + i);
    dst[i + 0] = f2bf(v.x);
    dst[i + 1] = f2bf(v.y);
    dst[i + 2] = f2bf(v.z);
    dst[i + 3] = f2bf(v.w);
  }
}

// ------------------------------------------- layernorm (+ optional delta GEMV)
__global__ __launch_bounds__(256) void k_layernorm(
    const float* __restrict__ src, const float* __restrict__ gam,
    const float* __restrict__ bet, bf16* __restrict__ dst_bf,
    bf16* __restrict__ raw_bf, const float* __restrict__ dw,
    const float* __restrict__ db, float* __restrict__ dout)
{
  __shared__ float s1[256];
  __shared__ float s2[256];
  const int row = blockIdx.x;
  const int tid = threadIdx.x;
  const float* p = src + (long)row * EDIM;
  v4f v = *(const v4f*)(p + tid * 4);
  s1[tid] = v.x + v.y + v.z + v.w;
  s2[tid] = v.x * v.x + v.y * v.y + v.z * v.z + v.w * v.w;
  __syncthreads();
  for (int off = 128; off > 0; off >>= 1) {
    if (tid < off) { s1[tid] += s1[tid + off]; s2[tid] += s2[tid + off]; }
    __syncthreads();
  }
  const float mean = s1[0] * (1.0f / EDIM);
  const float var  = s2[0] * (1.0f / EDIM) - mean * mean;
  const float rstd = rsqrtf(var + 1e-5f);
  float dot = 0.0f;
  #pragma unroll
  for (int j = 0; j < 4; ++j) {
    int e = tid * 4 + j;
    float xv = ((const float*)&v)[j];
    float xn = (xv - mean) * rstd * gam[e] + bet[e];
    dst_bf[(long)row * EDIM + e] = f2bf(xn);
    if (raw_bf) raw_bf[(long)row * EDIM + e] = f2bf(xv);
    if (dw) dot += xn * dw[e];
  }
  if (dw) {
    __syncthreads();
    s1[tid] = dot;
    __syncthreads();
    for (int off = 128; off > 0; off >>= 1) {
      if (tid < off) s1[tid] += s1[tid + off];
      __syncthreads();
    }
    if (tid == 0) {
      float z = s1[0] + db[0];
      dout[row] = (z > 20.0f) ? z : log1pf(__expf(z));   // softplus
    }
  }
}

// ------------------------------------------------ bf16 WMMA GEMM  C = A * B^T
// A: [M,K] row-major bf16, B: [N,K] row-major bf16, C: [M,N] f32
// mode 0: none   mode 1: sigmoid(acc + bias[n])   mode 2: gelu(acc + bias[n])
#define TBM 128
#define TBN 64
#define TBK 32
#define LDT 40

__global__ __launch_bounds__(256) void k_gemm_bf16(
    const bf16* __restrict__ A, const bf16* __restrict__ B,
    float* __restrict__ C, int M, int N, int K, int mode,
    const float* __restrict__ bias)
{
  __shared__ alignas(16) bf16 sA[2][TBM * LDT];
  __shared__ alignas(16) bf16 sB[2][TBN * LDT];
  const int tid   = threadIdx.x;
  const int nbase = blockIdx.x * TBN;
  const int mbase = blockIdx.y * TBM;
  const int wave  = tid >> 5;
  const int lane  = tid & 31;
  const int wm    = wave & 3;    // 4 waves along M
  const int wn    = wave >> 2;   // 2 waves along N
  const int lrow  = lane & 15;
  const int half  = lane >> 4;

  const int arow = tid >> 1;          // 128 rows, 2 threads/row
  const int acol = (tid & 1) * 16;
#if !HAVE_TDM
  const int brow = tid >> 2;          // 64 rows, 4 threads/row
  const int bcol = (tid & 3) * 8;
#endif

  auto stage = [&](int buf, int kc) {
    const bf16* ag = A + (long)(mbase + arow) * K + kc + acol;
#if GEMM_ASYNC
    async_ld_b128(ag,     &sA[buf][arow * LDT + acol]);
    async_ld_b128(ag + 8, &sA[buf][arow * LDT + acol + 8]);
#else
    *(v8bf*)&sA[buf][arow * LDT + acol]     = *(const v8bf*)ag;
    *(v8bf*)&sA[buf][arow * LDT + acol + 8] = *(const v8bf*)(ag + 8);
    if (kc + TBK < K) __builtin_prefetch((const void*)(ag + TBK), 0, 1);
#endif
#if HAVE_TDM
    if (wave == 0) {
      tdm_load_2d(B + (long)nbase * K + kc, lds_addr_of(&sB[buf][0]),
                  TBK, TBN, (unsigned long)K,
                  (unsigned)(K - kc), TBN, true);
    }
#else
    const bf16* bg = B + (long)(nbase + brow) * K + kc + bcol;
    *(v8bf*)&sB[buf][brow * LDT + bcol] = *(const v8bf*)bg;
    if (kc + TBK < K) __builtin_prefetch((const void*)(bg + TBK), 0, 1);
#endif
  };
  auto wait_stage = [&]() {
#if GEMM_ASYNC
    __builtin_amdgcn_s_wait_asynccnt(0);
#endif
#if HAVE_TDM
    if (wave == 0) __builtin_amdgcn_s_wait_tensorcnt(0);
#endif
    __syncthreads();
  };

  v8f acc[2][2] = {};
  int buf = 0;
  stage(0, 0);
  wait_stage();

  for (int kc = 0; kc < K; kc += TBK) {
    if (kc + TBK < K) stage(buf ^ 1, kc + TBK);   // overlap copy with compute

    FragBF a0, a1, b0, b1;
    const bf16* pa0 = &sA[buf][(wm * 32 +  0 + lrow) * LDT];
    const bf16* pa1 = &sA[buf][(wm * 32 + 16 + lrow) * LDT];
    const bf16* pb0 = &sB[buf][(wn * 32 +  0 + lrow) * LDT];
    const bf16* pb1 = &sB[buf][(wn * 32 + 16 + lrow) * LDT];
    const int k0 = 8 * half, k1 = 16 + 8 * half;   // two 8-elem K runs per lane
    a0.h[0] = *(const v8bf*)(pa0 + k0); a0.h[1] = *(const v8bf*)(pa0 + k1);
    a1.h[0] = *(const v8bf*)(pa1 + k0); a1.h[1] = *(const v8bf*)(pa1 + k1);
    b0.h[0] = *(const v8bf*)(pb0 + k0); b0.h[1] = *(const v8bf*)(pb0 + k1);
    b1.h[0] = *(const v8bf*)(pb1 + k0); b1.h[1] = *(const v8bf*)(pb1 + k1);

    acc[0][0] = __builtin_amdgcn_wmma_f32_16x16x32_bf16(false, a0.v, false, b0.v, (short)0, acc[0][0], false, false);
    acc[0][1] = __builtin_amdgcn_wmma_f32_16x16x32_bf16(false, a0.v, false, b1.v, (short)0, acc[0][1], false, false);
    acc[1][0] = __builtin_amdgcn_wmma_f32_16x16x32_bf16(false, a1.v, false, b0.v, (short)0, acc[1][0], false, false);
    acc[1][1] = __builtin_amdgcn_wmma_f32_16x16x32_bf16(false, a1.v, false, b1.v, (short)0, acc[1][1], false, false);

    wait_stage();
    buf ^= 1;
  }

  #pragma unroll
  for (int mt = 0; mt < 2; ++mt)
    #pragma unroll
    for (int nt = 0; nt < 2; ++nt)
      #pragma unroll
      for (int r = 0; r < 8; ++r) {
        int row = mbase + wm * 32 + mt * 16 + r + 8 * half;
        int col = nbase + wn * 32 + nt * 16 + lrow;
        float v = acc[mt][nt][r];
        if (mode == 1) { v += bias[col]; v = 1.0f / (1.0f + __expf(-v)); }
        else if (mode == 2) { v += bias[col]; v = 0.5f * v * (1.0f + erff(v * 0.70710678f)); }
        C[(long)row * N + col] = v;
      }
}

// -------------------------------------- reservoir scan (WMMA per step, cluster)
// 8 workgroups x 8 waves; wave owns 16 H columns, h-state lives in accumulators.
// W_res slice (128 x 1024 bf16 = 256 KB) staged once into dynamic LDS via TDM.
__global__ __launch_bounds__(256) void k_reservoir(
    const bf16* __restrict__ Wres, const float* __restrict__ xin,
    float* __restrict__ res, bf16* __restrict__ hbuf)
{
  extern __shared__ bf16 swres[];          // 128 * HDIM bf16
  const int tid  = threadIdx.x;
  const int lane = tid & 31;
  const int wave = tid >> 5;
  const int lrow = lane & 15;
  const int half = lane >> 4;
  const int rbase = blockIdx.x * 128;
  const int j0    = rbase + wave * 16;

  // stage the workgroup's W_res row-slice into LDS (once)
#if HAVE_TDM
  if (wave == 0) {
    tdm_load_2d(Wres + (long)rbase * HDIM, lds_addr_of(&swres[0]),
                HDIM, 128, (unsigned long)HDIM, HDIM, 128, false);
  }
#else
  {
    const v8bf* src = (const v8bf*)(Wres + (long)rbase * HDIM);
    for (int i = tid; i < 128 * HDIM / 8; i += 256) ((v8bf*)swres)[i] = src[i];
  }
#endif

  // zero both h double-buffers (h0 == 0)
  for (int i = blockIdx.x * 256 + tid; i < 2 * 16 * HDIM; i += 8 * 256)
    hbuf[i] = f2bf(0.0f);

#if HAVE_TDM
  if (wave == 0) __builtin_amdgcn_s_wait_tensorcnt(0);
#endif
  __syncthreads();
  __threadfence();
  __builtin_amdgcn_s_cluster_barrier();

  float hreg[8];
  #pragma unroll
  for (int r = 0; r < 8; ++r) hreg[r] = 0.0f;

  const int lds_row = wave * 16 + lrow;    // local row of this lane's B fragment
  for (int t = 0; t < SEQ; ++t) {
    const bf16* hcur = hbuf + (t & 1) * 16 * HDIM;
    bf16* hnxt       = hbuf + ((t + 1) & 1) * 16 * HDIM;
    v8f acc = {};
    const int k0 = 8 * half, k1 = 16 + 8 * half;
    for (int kc = 0; kc < HDIM; kc += 32) {
      FragBF a, b;
      const bf16* pa = hcur + lrow * HDIM + kc;          // A: h (16 x K), global/L2
      const bf16* pb = swres + (long)lds_row * HDIM + kc; // B: W_res rows from LDS
      a.h[0] = *(const v8bf*)(pa + k0); a.h[1] = *(const v8bf*)(pa + k1);
      b.h[0] = *(const v8bf*)(pb + k0); b.h[1] = *(const v8bf*)(pb + k1);
      acc = __builtin_amdgcn_wmma_f32_16x16x32_bf16(false, a.v, false, b.v, (short)0, acc, false, false);
    }
    #pragma unroll
    for (int r = 0; r < 8; ++r) {
      const int m = r + 8 * half;       // C layout: vgpr r, lane half -> row
      const int j = j0 + lrow;
      float hnew = 0.0f;
      if (m < BATCH) {
        long idx = ((long)m * SEQ + t) * HDIM + j;
        float u = xin[idx];
        hnew = 0.95f * tanhf(u + acc[r]) + 0.05f * hreg[r];
        res[idx] = hnew;
      }
      hreg[r] = hnew;
      hnxt[m * HDIM + j] = f2bf(hnew);
    }
    __syncthreads();
    __threadfence();
    __builtin_amdgcn_s_cluster_barrier();
  }
}

// -------------------------------------------------- selective-SSM channel scan
__global__ __launch_bounds__(256) void k_ssm(
    const float* __restrict__ Avec, const float* __restrict__ delta,
    const float* __restrict__ Bx, float* __restrict__ ssm)
{
  int gid = blockIdx.x * 256 + threadIdx.x;   // 8192 channels
  int b = gid >> 10, h = gid & 1023;
  float a = Avec[h];
  float inv = 1.0f / (a + 1e-8f);
  float s = 0.0f;
  const float* dp = delta + (long)b * SEQ;
  for (int t = 0; t < SEQ; ++t) {
    float ab = __expf(a * dp[t]);
    long idx = ((long)b * SEQ + t) * HDIM + h;
    s = ab * s + Bx[idx] * (1.0f - ab) * inv;
    ssm[idx] = s;
  }
}

// -------------------------------------------------------------- gate blend
__global__ __launch_bounds__(256) void k_combine(
    float* __restrict__ out, const float* __restrict__ gate,
    const float* __restrict__ ssm)
{
  long i = ((long)blockIdx.x * 256 + threadIdx.x) * 4;
  v4f o = *(const v4f*)(out + i);
  v4f g = *(const v4f*)(gate + i);
  v4f s = *(const v4f*)(ssm + i);
  v4f r;
  r.x = o.x * g.x + s.x * (1.0f - g.x);
  r.y = o.y * g.y + s.y * (1.0f - g.y);
  r.z = o.z * g.z + s.z * (1.0f - g.z);
  r.w = o.w * g.w + s.w * (1.0f - g.w);
  *(v4f*)(out + i) = r;
}

// ================================================================= launcher
extern "C" void kernel_launch(void* const* d_in, const int* in_sizes, int n_in,
                              void* d_out, int out_size, void* d_ws, size_t ws_size,
                              hipStream_t stream) {
  const float* x       = (const float*)d_in[0];
  const float* W_in    = (const float*)d_in[1];
  const float* W_res   = (const float*)d_in[2];
  const float* norm_g  = (const float*)d_in[3];
  const float* norm_b  = (const float*)d_in[4];
  const float* delta_w = (const float*)d_in[5];
  const float* delta_b = (const float*)d_in[6];
  const float* Avec    = (const float*)d_in[7];
  const float* B_w     = (const float*)d_in[8];
  const float* ro_g    = (const float*)d_in[9];
  const float* ro_b    = (const float*)d_in[10];
  const float* ro_w    = (const float*)d_in[11];
  const float* ro_bias = (const float*)d_in[12];
  const float* gate_w  = (const float*)d_in[13];
  const float* gate_b  = (const float*)d_in[14];
  float* out = (float*)d_out;

  char* p = (char*)d_ws;
  auto alloc = [&](size_t bytes) -> char* {
    char* r = p; p += (bytes + 255) & ~(size_t)255; return r;
  };
  bf16* x_bf    = (bf16*)alloc((size_t)ROWS * EDIM * 2);
  bf16* xn_bf   = (bf16*)alloc((size_t)ROWS * EDIM * 2);
  bf16* roin_bf = (bf16*)alloc((size_t)ROWS * HDIM * 2);
  bf16* wbf     = (bf16*)alloc((size_t)HDIM * EDIM * 2);
  bf16* bwbf    = (bf16*)alloc((size_t)HDIM * EDIM * 2);
  bf16* gwbf    = (bf16*)alloc((size_t)EDIM * EDIM * 2);
  bf16* rowbf   = (bf16*)alloc((size_t)EDIM * HDIM * 2);
  bf16* wrbf    = (bf16*)alloc((size_t)HDIM * HDIM * 2);
  bf16* hbuf    = (bf16*)alloc((size_t)2 * 16 * HDIM * 2);
  float* delta  = (float*)alloc((size_t)ROWS * 4);
  float* xin    = (float*)alloc((size_t)ROWS * HDIM * 4);
  float* Bx     = (float*)alloc((size_t)ROWS * HDIM * 4);
  float* gate   = (float*)alloc((size_t)ROWS * EDIM * 4);
  float* res    = (float*)alloc((size_t)ROWS * HDIM * 4);
  float* ssm    = (float*)alloc((size_t)ROWS * HDIM * 4);
  if ((size_t)(p - (char*)d_ws) > ws_size) return;

  // 1. weight casts to bf16 (L2 resident thereafter)
  auto cvt = [&](const float* s, bf16* d, long n) {
    k_f32_to_bf16<<<dim3((unsigned)(n / 1024)), dim3(256), 0, stream>>>(s, d, (int)n);
  };
  cvt(W_in,   wbf,   (long)HDIM * EDIM);
  cvt(B_w,    bwbf,  (long)HDIM * EDIM);
  cvt(gate_w, gwbf,  (long)EDIM * EDIM);
  cvt(ro_w,   rowbf, (long)EDIM * HDIM);
  cvt(W_res,  wrbf,  (long)HDIM * HDIM);

  // 2. layernorm + bf16 casts + delta softplus (fused)
  k_layernorm<<<dim3(ROWS), dim3(256), 0, stream>>>(
      x, norm_g, norm_b, xn_bf, x_bf, delta_w, delta_b, delta);

  // 3. the three input-side GEMMs
  dim3 gg(HDIM / TBN, ROWS / TBM);   // (16, 128)
  k_gemm_bf16<<<gg, dim3(256), 0, stream>>>(xn_bf, wbf,  xin,  ROWS, HDIM, EDIM, 0, nullptr);
  k_gemm_bf16<<<gg, dim3(256), 0, stream>>>(xn_bf, bwbf, Bx,   ROWS, HDIM, EDIM, 0, nullptr);
  k_gemm_bf16<<<gg, dim3(256), 0, stream>>>(x_bf,  gwbf, gate, ROWS, EDIM, EDIM, 1, gate_b);

  // 4. reservoir recurrence (cluster-synced WMMA scan, W_res slice in LDS)
  k_reservoir<<<dim3(8), dim3(256), 128 * HDIM * sizeof(bf16), stream>>>(
      wrbf, xin, res, hbuf);

  // 5. readout layernorm + GELU GEMM straight into d_out
  k_layernorm<<<dim3(ROWS), dim3(256), 0, stream>>>(
      res, ro_g, ro_b, roin_bf, nullptr, nullptr, nullptr, nullptr);
  k_gemm_bf16<<<gg, dim3(256), 0, stream>>>(roin_bf, rowbf, out, ROWS, EDIM, HDIM, 2, ro_bias);

  // 6. SSM scan + gate blend
  k_ssm<<<dim3((BATCH * HDIM) / 256), dim3(256), 0, stream>>>(Avec, delta, Bx, ssm);
  k_combine<<<dim3((unsigned)((long)ROWS * EDIM / 1024)), dim3(256), 0, stream>>>(out, gate, ssm);
}
